// HeterogeneousGNN_46995532153265
// MI455X (gfx1250) — compile-verified, matrix operand order checked by reading, not run on hardware
//
#include <hip/hip_runtime.h>
#include <hip/hip_bf16.h>

// ---------------------------------------------------------------------------
// HeterogeneousGNN for MI455X (gfx1250, wave32, WMMA)
// ---------------------------------------------------------------------------
#define NN   50000
#define EE   250000
#define HH   128
#define EPSV 1e-5f

typedef __attribute__((ext_vector_type(16))) _Float16 v16h;
typedef __attribute__((ext_vector_type(8)))  _Float16 v8h;
typedef __attribute__((ext_vector_type(8)))  float    v8f;

__device__ __forceinline__ void atomAddF(float* p, float v) {
    unsafeAtomicAdd(p, v);   // native global_atomic_add_f32 / ds_add_f32
}

// Load one 16x32 f16 A/B fragment row-slice for this lane from row-major data.
// base = start of this lane's row; halves [kbase..kbase+7] and [kbase+16..kbase+23].
__device__ __forceinline__ v16h load_frag(const _Float16* __restrict__ base, int kbase) {
    const v8h g0 = *(const v8h*)(base + kbase);
    const v8h g1 = *(const v8h*)(base + kbase + 16);
    v16h r;
#pragma unroll
    for (int i = 0; i < 8; ++i) { r[i] = g0[i]; r[i + 8] = g1[i]; }
    return r;
}

// ---------------------------------------------------------------------------
// Small utility kernels
// ---------------------------------------------------------------------------
__global__ void fill1_k(float* p, int n) {
    int i = blockIdx.x * blockDim.x + threadIdx.x;
    if (i < n) p[i] = 1.0f;
}

__global__ void deg_scatter_k(const int* __restrict__ dst, float* deg, int nE) {
    int e = blockIdx.x * blockDim.x + threadIdx.x;
    if (e < nE) atomAddF(&deg[dst[e]], 1.0f);
}

__global__ void rsq_k(float* p, int n) {
    int i = blockIdx.x * blockDim.x + threadIdx.x;
    if (i < n) p[i] = rsqrtf(p[i]);
}

// Transpose + convert weight W[K][Nn] (f32, row-major) -> Wt[Nn][K] (f16)
__global__ void wt_cvt_k(const float* __restrict__ W, _Float16* __restrict__ Wt,
                         int K, int Nn) {
    int idx = blockIdx.x * blockDim.x + threadIdx.x;
    if (idx >= K * Nn) return;
    int n = idx / K, k = idx - n * K;
    Wt[idx] = (_Float16)W[(size_t)k * Nn + n];
}

// Node encoder: y = relu(x @ W_enc + b), accumulate per-column sum/sumsq
__global__ __launch_bounds__(128) void encoder_k(
    const float* __restrict__ x, const float* __restrict__ W,
    const float* __restrict__ b, float* __restrict__ y,
    float* sum, float* sumsq) {
    int h = threadIdx.x;
    float w0 = W[h], w1 = W[HH + h], w2 = W[2 * HH + h], bb = b[h];
    float s = 0.f, s2 = 0.f;
    for (int r = blockIdx.x; r < NN; r += gridDim.x) {
        float v = fmaxf(x[r * 3] * w0 + x[r * 3 + 1] * w1 + x[r * 3 + 2] * w2 + bb, 0.f);
        y[(size_t)r * HH + h] = v;
        s += v; s2 += v * v;
    }
    atomAddF(&sum[h], s);
    atomAddF(&sumsq[h], s2);
}

// BatchNorm finalize: a = g*rsqrt(var+eps), c = beta - mean*a
__global__ void bn_finalize_k(const float* sum, const float* sumsq,
                              const float* g, const float* be,
                              float* a, float* c, float count) {
    int h = threadIdx.x;
    float m = sum[h] / count;
    float v = sumsq[h] / count - m * m;
    float ai = g[h] * rsqrtf(v + EPSV);
    a[h] = ai;
    c[h] = be[h] - m * ai;
}

// Apply BN (per-column affine) and convert to f16
__global__ void bn_apply_f16_k(const float* __restrict__ y,
                               const float* __restrict__ a,
                               const float* __restrict__ c,
                               _Float16* __restrict__ out, int total) {
    int i = blockIdx.x * blockDim.x + threadIdx.x;
    if (i >= total) return;
    int h = i & (HH - 1);
    out[i] = (_Float16)(a[h] * y[i] + c[h]);
}

// Optional ReLU then f32 -> f16
__global__ void relu_cvt_k(const float* __restrict__ in, _Float16* __restrict__ out,
                           int total, int do_relu) {
    int i = blockIdx.x * blockDim.x + threadIdx.x;
    if (i >= total) return;
    float v = in[i];
    if (do_relu) v = fmaxf(v, 0.f);
    out[i] = (_Float16)v;
}

// agg = b_b + b_c + dinvb^2 * xwb + dinvc^2 * xwc   (self-loop terms + biases)
__global__ void agg_init_k(const float* __restrict__ xwb, const float* __restrict__ xwc,
                           const float* __restrict__ dinvb, const float* __restrict__ dinvc,
                           const float* __restrict__ bb, const float* __restrict__ bc,
                           float* __restrict__ agg, int total) {
    int i = blockIdx.x * blockDim.x + threadIdx.x;
    if (i >= total) return;
    int n = i >> 7, h = i & (HH - 1);
    float db = dinvb[n], dc = dinvc[n];
    agg[i] = bb[h] + bc[h] + db * db * xwb[i] + dc * dc * xwc[i];
}

// Scatter: agg[dst] += dinv[src]*dinv[dst] * xw[src]   (4 h per thread)
__global__ void gcn_scatter_k(const int* __restrict__ src, const int* __restrict__ dst,
                              const float* __restrict__ dinv, const float* __restrict__ xw,
                              float* __restrict__ agg, int nE) {
    int idx = blockIdx.x * blockDim.x + threadIdx.x;
    if (idx >= nE * 32) return;
    int e = idx >> 5;
    int h4 = (idx & 31) << 2;
    int s = src[e], d = dst[e];
    float norm = dinv[s] * dinv[d];
    const float4 v = *(const float4*)(xw + (size_t)s * HH + h4);
    float* o = agg + (size_t)d * HH + h4;
    atomAddF(o + 0, norm * v.x);
    atomAddF(o + 1, norm * v.y);
    atomAddF(o + 2, norm * v.z);
    atomAddF(o + 3, norm * v.w);
}

// edge_types output
__global__ void types_k(float* out, int n2E, int nE) {
    int i = blockIdx.x * blockDim.x + threadIdx.x;
    if (i < n2E) out[i] = (i < nE) ? 0.f : 1.f;
}

// ---------------------------------------------------------------------------
// Persistent WMMA GEMM: C[M,128] = A[M,128](f16) x W (f16, stored Wt[128][128]).
// 8 waves/block cover the 128 output cols; weight fragments hoisted into
// registers once per wave; A/C pointers marched so the loop body is pure
// load -> wmma -> store-with-immediate-offset.
// ---------------------------------------------------------------------------
__global__ __launch_bounds__(256) void gemm_node_k(
    const _Float16* __restrict__ A, const _Float16* __restrict__ Wt,
    float* __restrict__ C, int mtiles) {
    const int lane = threadIdx.x & 31;
    const int wave = threadIdx.x >> 5;
    const int n0 = wave << 4;
    const int r  = lane & 15;
    const int hs = lane >> 4;
    const _Float16* brow = Wt + (size_t)(n0 + r) * HH;
    v16h bf[4];
#pragma unroll
    for (int i = 0; i < 4; ++i) bf[i] = load_frag(brow, 32 * i + hs * 8);

    const size_t step = (size_t)gridDim.x * 16 * HH;   // elements per grid stride
    const _Float16* arow = A + (size_t)(blockIdx.x * 16 + r) * HH;
    float*          crow = C + (size_t)(blockIdx.x * 16 + hs * 8) * HH + n0 + r;
    for (int t = blockIdx.x; t < mtiles; t += gridDim.x) {
        v8f acc = {};
#pragma unroll
        for (int i = 0; i < 4; ++i) {
            v16h a = load_frag(arow, 32 * i + hs * 8);
            acc = __builtin_amdgcn_wmma_f32_16x16x32_f16(false, a, false, bf[i],
                                                         (short)0, acc, false, false);
        }
#pragma unroll
        for (int j = 0; j < 8; ++j) crow[j * HH] = acc[j];   // immediate offsets
        arow += step;
        crow += step;
    }
}

// ---------------------------------------------------------------------------
// Edge-head hidden tile with pre-hoisted weight fragments.
// K=256 = [h2[src] | h2[dst]]; bf[8] holds this wave's Wt[n0..n0+15][0..255].
// ---------------------------------------------------------------------------
__device__ __forceinline__ v8f edge_hidden_tile(
    const _Float16* __restrict__ h2, const int* __restrict__ src,
    const int* __restrict__ dst, const v16h* bf, int e0, int lane) {
    const int r  = lane & 15;
    const int hs = lane >> 4;
    const int e  = e0 + r;
    const _Float16* asrc = h2 + (size_t)src[e] * HH;
    const _Float16* adst = h2 + (size_t)dst[e] * HH;
    v8f acc = {};
#pragma unroll
    for (int i = 0; i < 8; ++i) {
        const int kk = i * 32;
        const _Float16* abase = (kk < HH) ? asrc : (adst - HH);
        v16h a = load_frag(abase, kk + hs * 8);
        acc = __builtin_amdgcn_wmma_f32_16x16x32_f16(false, a, false, bf[i],
                                                     (short)0, acc, false, false);
    }
    return acc;
}

__device__ __forceinline__ void load_edge_bfrags(
    const _Float16* __restrict__ Wt, int n0, int lane, v16h* bf) {
    const int r  = lane & 15;
    const int hs = lane >> 4;
    const _Float16* brow = Wt + (size_t)(n0 + r) * 256;
#pragma unroll
    for (int i = 0; i < 8; ++i) bf[i] = load_frag(brow, 32 * i + hs * 8);
}

// Pass 1: BN statistics over hidden = relu(edge_feat @ W1 + b1); grid-stride.
__global__ __launch_bounds__(256) void edge_stats_k(
    const _Float16* __restrict__ h2, const int* __restrict__ src,
    const int* __restrict__ dst, const _Float16* __restrict__ Wt,
    const float* __restrict__ bias, float* sum, float* sumsq, int ntiles) {
    const int lane = threadIdx.x & 31;
    const int wave = threadIdx.x >> 5;
    const int n0 = wave << 4;
    const int col = n0 + (lane & 15);
    const float bcol = bias[col];
    v16h bf[8];
    load_edge_bfrags(Wt, n0, lane, bf);
    float s = 0.f, s2 = 0.f;
    for (int t = blockIdx.x; t < ntiles; t += gridDim.x) {
        v8f acc = edge_hidden_tile(h2, src, dst, bf, t << 4, lane);
#pragma unroll
        for (int j = 0; j < 8; ++j) {
            float v = fmaxf(acc[j] + bcol, 0.f);
            s += v; s2 += v * v;
        }
    }
    float so  = __shfl(s,  lane + 16);
    float s2o = __shfl(s2, lane + 16);
    if (lane < 16) {
        atomAddF(&sum[col],   s  + so);
        atomAddF(&sumsq[col], s2 + s2o);
    }
}

// Pass 2: recompute hidden, apply BN, project 128->2, write [E,2] outputs.
// Persistent: weight fragments hoisted, grid-stride over edge tiles.
__global__ __launch_bounds__(256) void edge_out_k(
    const _Float16* __restrict__ h2, const int* __restrict__ src,
    const int* __restrict__ dst, const _Float16* __restrict__ Wt,
    const float* __restrict__ bias, const float* __restrict__ bn_a,
    const float* __restrict__ bn_c, const float* __restrict__ W2,
    const float* __restrict__ b2, float* __restrict__ out, int ntiles) {
    __shared__ float lds[16][2];
    const int lane = threadIdx.x & 31;
    const int wave = threadIdx.x >> 5;
    const int n0 = wave << 4;
    const int col = n0 + (lane & 15);
    const int hs = lane >> 4;
    const float bcol = bias[col], a = bn_a[col], c = bn_c[col];
    const float w0 = W2[col * 2 + 0], w1 = W2[col * 2 + 1];
    const float ob0 = b2[0], ob1 = b2[1];
    v16h bf[8];
    load_edge_bfrags(Wt, n0, lane, bf);
    for (int t = blockIdx.x; t < ntiles; t += gridDim.x) {
        const int e0 = t << 4;
        if (threadIdx.x < 32) lds[threadIdx.x >> 1][threadIdx.x & 1] = 0.f;
        __syncthreads();
        v8f acc = edge_hidden_tile(h2, src, dst, bf, e0, lane);
#pragma unroll
        for (int j = 0; j < 8; ++j) {
            float v = fmaxf(acc[j] + bcol, 0.f);
            v = a * v + c;
            float p0 = v * w0, p1 = v * w1;
#pragma unroll
            for (int m = 1; m < 16; m <<= 1) {   // reduce 16 cols per half-wave
                p0 += __shfl_xor(p0, m);
                p1 += __shfl_xor(p1, m);
            }
            if ((lane & 15) == 0) {
                int row = j + hs * 8;
                atomAddF(&lds[row][0], p0);
                atomAddF(&lds[row][1], p1);
            }
        }
        __syncthreads();
        if (threadIdx.x < 32) {
            int row = threadIdx.x >> 1, d = threadIdx.x & 1;
            out[(size_t)(e0 + row) * 2 + d] = lds[row][d] + (d ? ob1 : ob0);
        }
        __syncthreads();
    }
}

// ---------------------------------------------------------------------------
extern "C" void kernel_launch(void* const* d_in, const int* in_sizes, int n_in,
                              void* d_out, int out_size, void* d_ws, size_t ws_size,
                              hipStream_t stream) {
    (void)in_sizes; (void)n_in; (void)out_size; (void)ws_size;
    const float* x      = (const float*)d_in[0];
    const int*   beam   = (const int*)d_in[1];
    const int*   coli   = (const int*)d_in[2];
    const float* W_enc  = (const float*)d_in[3];
    const float* b_enc  = (const float*)d_in[4];
    const float* g_enc  = (const float*)d_in[5];
    const float* be_enc = (const float*)d_in[6];
    const float* W1b = (const float*)d_in[7];  const float* b1b = (const float*)d_in[8];
    const float* W1c = (const float*)d_in[9];  const float* b1c = (const float*)d_in[10];
    const float* W2b = (const float*)d_in[11]; const float* b2b = (const float*)d_in[12];
    const float* W2c = (const float*)d_in[13]; const float* b2c = (const float*)d_in[14];
    const float* Wbp1 = (const float*)d_in[15]; const float* bbp1 = (const float*)d_in[16];
    const float* gbp  = (const float*)d_in[17]; const float* bebp = (const float*)d_in[18];
    const float* Wbp2 = (const float*)d_in[19]; const float* bbp2 = (const float*)d_in[20];
    const float* Wcp1 = (const float*)d_in[21]; const float* bcp1 = (const float*)d_in[22];
    const float* gcp  = (const float*)d_in[23]; const float* becp = (const float*)d_in[24];
    const float* Wcp2 = (const float*)d_in[25]; const float* bcp2 = (const float*)d_in[26];

    const int* bs = beam;       const int* bd = beam + EE;
    const int* cs = coli;       const int* cd = coli + EE;

    // ---- workspace carve-out ----
    char* ws = (char*)d_ws;
    size_t off = 0;
    auto take = [&](size_t bytes) -> char* {
        char* p = ws + off;
        off = (off + bytes + 255) & ~(size_t)255;
        return p;
    };
    float*     stats = (float*)take(16 * HH * sizeof(float));
    float*     dinvb = (float*)take((size_t)NN * sizeof(float));
    float*     dinvc = (float*)take((size_t)NN * sizeof(float));
    float*     bufA  = (float*)take((size_t)NN * HH * sizeof(float));
    float*     bufB  = (float*)take((size_t)NN * HH * sizeof(float));
    float*     bufC  = (float*)take((size_t)NN * HH * sizeof(float));
    _Float16*  hF    = (_Float16*)take((size_t)NN * HH * 2);
    _Float16*  h1F   = (_Float16*)take((size_t)NN * HH * 2);
    _Float16*  h2F   = (_Float16*)take((size_t)NN * HH * 2);
    _Float16*  wt1b  = (_Float16*)take((size_t)HH * HH * 2);
    _Float16*  wt1c  = (_Float16*)take((size_t)HH * HH * 2);
    _Float16*  wt2b  = (_Float16*)take((size_t)HH * HH * 2);
    _Float16*  wt2c  = (_Float16*)take((size_t)HH * HH * 2);
    _Float16*  wtbp  = (_Float16*)take((size_t)HH * 256 * 2);
    _Float16*  wtcp  = (_Float16*)take((size_t)HH * 256 * 2);

    float* enc_sum = stats;           float* enc_sq = stats + 128;
    float* enc_a   = stats + 256;     float* enc_c  = stats + 384;
    float* bp_sum  = stats + 512;     float* bp_sq  = stats + 640;
    float* bp_a    = stats + 768;     float* bp_c   = stats + 896;
    float* cp_sum  = stats + 1024;    float* cp_sq  = stats + 1152;
    float* cp_a    = stats + 1280;    float* cp_c   = stats + 1408;

    hipMemsetAsync(stats, 0, 16 * HH * sizeof(float), stream);

    const int T = 256;
    const int NH = NN * HH;
    const int mtiles_n = NN / 16;            // 3125
    const int ntiles_e = EE / 16;            // 15625
    float* out_beam = (float*)d_out;
    float* out_col  = out_beam + (size_t)EE * 2;
    float* out_ty   = out_beam + (size_t)EE * 4;

    // weights -> f16, transposed to [N][K]
    wt_cvt_k<<<(HH * HH + T - 1) / T, T, 0, stream>>>(W1b,  wt1b, HH,  HH);
    wt_cvt_k<<<(HH * HH + T - 1) / T, T, 0, stream>>>(W1c,  wt1c, HH,  HH);
    wt_cvt_k<<<(HH * HH + T - 1) / T, T, 0, stream>>>(W2b,  wt2b, HH,  HH);
    wt_cvt_k<<<(HH * HH + T - 1) / T, T, 0, stream>>>(W2c,  wt2c, HH,  HH);
    wt_cvt_k<<<(256 * HH + T - 1) / T, T, 0, stream>>>(Wbp1, wtbp, 256, HH);
    wt_cvt_k<<<(256 * HH + T - 1) / T, T, 0, stream>>>(Wcp1, wtcp, 256, HH);

    // degrees -> dinv = rsqrt(indeg + 1)
    fill1_k<<<(NN + T - 1) / T, T, 0, stream>>>(dinvb, NN);
    fill1_k<<<(NN + T - 1) / T, T, 0, stream>>>(dinvc, NN);
    deg_scatter_k<<<(EE + T - 1) / T, T, 0, stream>>>(bd, dinvb, EE);
    deg_scatter_k<<<(EE + T - 1) / T, T, 0, stream>>>(cd, dinvc, EE);
    rsq_k<<<(NN + T - 1) / T, T, 0, stream>>>(dinvb, NN);
    rsq_k<<<(NN + T - 1) / T, T, 0, stream>>>(dinvc, NN);

    // encoder + BN -> h (f16)
    encoder_k<<<512, 128, 0, stream>>>(x, W_enc, b_enc, bufA, enc_sum, enc_sq);
    bn_finalize_k<<<1, 128, 0, stream>>>(enc_sum, enc_sq, g_enc, be_enc, enc_a, enc_c, (float)NN);
    bn_apply_f16_k<<<(NH + T - 1) / T, T, 0, stream>>>(bufA, enc_a, enc_c, hF, NH);

    // GCN layer 1
    gemm_node_k<<<640, T, 0, stream>>>(hF, wt1b, bufB, mtiles_n);
    gemm_node_k<<<640, T, 0, stream>>>(hF, wt1c, bufC, mtiles_n);
    agg_init_k<<<(NH + T - 1) / T, T, 0, stream>>>(bufB, bufC, dinvb, dinvc, b1b, b1c, bufA, NH);
    gcn_scatter_k<<<(EE * 32 + T - 1) / T, T, 0, stream>>>(bs, bd, dinvb, bufB, bufA, EE);
    gcn_scatter_k<<<(EE * 32 + T - 1) / T, T, 0, stream>>>(cs, cd, dinvc, bufC, bufA, EE);
    relu_cvt_k<<<(NH + T - 1) / T, T, 0, stream>>>(bufA, h1F, NH, 1);

    // GCN layer 2
    gemm_node_k<<<640, T, 0, stream>>>(h1F, wt2b, bufB, mtiles_n);
    gemm_node_k<<<640, T, 0, stream>>>(h1F, wt2c, bufC, mtiles_n);
    agg_init_k<<<(NH + T - 1) / T, T, 0, stream>>>(bufB, bufC, dinvb, dinvc, b2b, b2c, bufA, NH);
    gcn_scatter_k<<<(EE * 32 + T - 1) / T, T, 0, stream>>>(bs, bd, dinvb, bufB, bufA, EE);
    gcn_scatter_k<<<(EE * 32 + T - 1) / T, T, 0, stream>>>(cs, cd, dinvc, bufC, bufA, EE);
    relu_cvt_k<<<(NH + T - 1) / T, T, 0, stream>>>(bufA, h2F, NH, 0);

    // beam edge head (stats pass -> BN finalize -> output pass)
    edge_stats_k<<<1280, T, 0, stream>>>(h2F, bs, bd, wtbp, bbp1, bp_sum, bp_sq, ntiles_e);
    bn_finalize_k<<<1, 128, 0, stream>>>(bp_sum, bp_sq, gbp, bebp, bp_a, bp_c, (float)EE);
    edge_out_k<<<1280, T, 0, stream>>>(h2F, bs, bd, wtbp, bbp1, bp_a, bp_c, Wbp2, bbp2,
                                       out_beam, ntiles_e);

    // col edge head
    edge_stats_k<<<1280, T, 0, stream>>>(h2F, cs, cd, wtcp, bcp1, cp_sum, cp_sq, ntiles_e);
    bn_finalize_k<<<1, 128, 0, stream>>>(cp_sum, cp_sq, gcp, becp, cp_a, cp_c, (float)EE);
    edge_out_k<<<1280, T, 0, stream>>>(h2F, cs, cd, wtcp, bcp1, cp_a, cp_c, Wcp2, bcp2,
                                       out_col, ntiles_e);

    // edge types
    types_k<<<(2 * EE + T - 1) / T, T, 0, stream>>>(out_ty, 2 * EE, EE);
}
